// JointNetwork_55740085567620
// MI455X (gfx1250) — matrix operand order repaired and checked by reading, not compiled, not run on hardware
//
#include <hip/hip_runtime.h>
#include <hip/hip_bf16.h>

// ---------------------------------------------------------------------------
// RNN-T joint network, fused for MI455X (gfx1250, wave32, WMMA, async-LDS).
//   Wt_*     = bf16 transposed copies of W_enc/W_dec/W_out         (once, ws)
//   enc_proj = enc_out @ W_enc + b_enc        (2048 x 512)          -> ws
//   dec_proj = dec_out @ W_dec + b_dec        ( 512 x 512)          -> ws
//   out      = tanh(enc_proj[bt] + dec_proj[bu]) @ W_out + b_out
// The (B,T,U,J) joint tensor (537 MB f32) is never materialized; each 64x32
// bf16 A-tile is synthesized in LDS from the two small projections.
// B tiles stream global->LDS with GLOBAL_LOAD_ASYNC_TO_LDS_B128 (ASYNCcnt),
// double-buffered. Matrix core: v_wmma_f32_16x16x32_bf16.
// ---------------------------------------------------------------------------

typedef __attribute__((ext_vector_type(16))) __bf16 v16bf;
typedef __attribute__((ext_vector_type(8)))  __bf16 v8bf;
typedef __attribute__((ext_vector_type(8)))  float  v8f;
typedef __attribute__((ext_vector_type(4)))  int    v4i;

#define B_DIM 4
#define T_DIM 512
#define U_DIM 128
#define J_DIM 512
#define V_DIM 1024
#define K_DIM 512      // == D_ENC == D_DEC == J

#define AST 40         // LDS row stride (elems) for A tile (64 x 32 + pad)
#define BST 40         // LDS row stride (elems) for B tile (transposed, N x K)

#if __has_builtin(__builtin_amdgcn_global_load_async_to_lds_b128)
#define HAVE_ASYNC_LDS 1
#else
#define HAVE_ASYNC_LDS 0
#endif

typedef __attribute__((address_space(1))) v4i* gv4i_ptr;   // global int4*
typedef __attribute__((address_space(3))) v4i* lv4i_ptr;   // LDS int4*

__device__ __forceinline__ void copy16_g2l(const void* g, void* l) {
#if HAVE_ASYNC_LDS
  __builtin_amdgcn_global_load_async_to_lds_b128(
      (gv4i_ptr)(void*)(size_t)g, (lv4i_ptr)l, 0, 0);
#else
  *(uint4*)l = *(const uint4*)g;
#endif
}

template <int N>
__device__ __forceinline__ void wait_async() {
#if HAVE_ASYNC_LDS
#if __has_builtin(__builtin_amdgcn_s_wait_asynccnt)
  __builtin_amdgcn_s_wait_asynccnt(N);
#else
  asm volatile("s_wait_asynccnt %0" ::"i"(N) : "memory");
#endif
#endif
}

__device__ __forceinline__ float fast_tanh(float x) {
#if __has_builtin(__builtin_amdgcn_tanhf)
  return __builtin_amdgcn_tanhf(x);   // native v_tanh_f32 on gfx1250
#else
  // tanh(x) = 1 - 2/(exp(2x)+1); v_exp_f32 + v_rcp_f32 + fma.
  // exp(2x) = exp2(x * 2*log2(e)). Saturates to +/-1 for large |x|.
  float e = __builtin_amdgcn_exp2f(x * 2.885390081777927f);
  return 1.0f - 2.0f * __builtin_amdgcn_rcpf(e + 1.0f);
#endif
}

__device__ __forceinline__ v16bf cat16(v8bf lo, v8bf hi) {
  return __builtin_shufflevector(lo, hi, 0, 1, 2, 3, 4, 5, 6, 7,
                                         8, 9, 10, 11, 12, 13, 14, 15);
}

// ---------------------------------------------------------------------------
// Weight prep: W (K_DIM x N, f32, row major) -> Wt (N x K_DIM, bf16).
// Consecutive threads write consecutive k for one n (coalesced 2B stores);
// reads gather from L2-resident W (4 MB total across all three weights).
// ---------------------------------------------------------------------------
__global__ __launch_bounds__(256) void wcvt(const float* __restrict__ W,
                                            __bf16* __restrict__ Wt, int N) {
  const int idx = blockIdx.x * 256 + threadIdx.x;  // over N * K_DIM
  const int n = idx >> 9;
  const int k = idx & (K_DIM - 1);
  Wt[idx] = (__bf16)W[(size_t)k * N + n];
}

// ---------------------------------------------------------------------------
// MODE 0: fused joint GEMM  (A = tanh(encp+decp), B = Wt_out, NCOLS = 1024)
// MODE 1: projection GEMM   (A = f32 source,      B = Wt_*,   NCOLS = 512)
// Block: 256 threads (8 wave32), tile 64M x 512N, K stepped by 32.
// ---------------------------------------------------------------------------
template <int MODE, int NCOLS>
__global__ __launch_bounds__(256)
void joint_gemm(const float* __restrict__ A0,     // MODE0: enc_proj, MODE1: A src
                const float* __restrict__ A1,     // MODE0: dec_proj
                const __bf16* __restrict__ Wt,    // (NCOLS x K_DIM) bf16
                const float* __restrict__ bias,   // (NCOLS)
                float* __restrict__ Cout) {       // (M x NCOLS)
  __shared__ __align__(16) __bf16 As[64 * AST];
  __shared__ __align__(16) __bf16 Bs[2][512 * BST];

  const int tid  = threadIdx.x;
  const int lane = tid & 31;
  const int wid  = tid >> 5;
  const int wm   = wid >> 2;                 // 0..1 : 32-row strip
  const int wn   = wid & 3;                  // 0..3 : 128-col strip
  const int m0   = blockIdx.x * 64;
  const int n0   = blockIdx.y * 512;

  int bt = 0, decbase = 0;
  if constexpr (MODE == 0) {
    bt = m0 >> 7;                            // row block shares (b,t): U=128, 64|m0
    const int b = bt >> 9;                   // T = 512
    decbase = b * U_DIM + (m0 & (U_DIM - 1));
  }

  v8f acc[2][8] = {};

  const int ar = tid >> 2;                   // 0..63   A-tile row this thread fills
  const int as = (tid & 3) * 8;              // 0,8,16,24  k-segment

  // B-tile streamer: 512 rows x 64B per K-step = 2048 x 16B chunks, 8/thread.
  auto fill_B = [&](int k0, __bf16* BsBuf) {
#pragma unroll
    for (int j = 0; j < 8; ++j) {
      const int id = tid + j * 256;
      const int n  = id >> 2;
      const int c  = (id & 3) * 16;          // byte offset within the 64B row
      const char* gs = (const char*)(Wt + (size_t)(n0 + n) * K_DIM + k0) + c;
      char*       ls = (char*)(BsBuf + n * BST) + c;
      copy16_g2l(gs, ls);
    }
  };

  fill_B(0, Bs[0]);                          // prime the pipeline

  constexpr int NIT = K_DIM / 32;            // 16
  for (int it = 0; it < NIT; ++it) {
    const int k0  = it * 32;
    const int buf = it & 1;

    // ---- A tile values (global f32 loads overlap previous WMMA block) ----
    v8bf av;
    if constexpr (MODE == 0) {
      const float4* pe = (const float4*)(A0 + (size_t)bt * J_DIM + k0 + as);
      const float4* pd =
          (const float4*)(A1 + (size_t)(decbase + ar) * J_DIM + k0 + as);
      float4 e0 = pe[0], e1 = pe[1], d0 = pd[0], d1 = pd[1];
      av[0] = (__bf16)fast_tanh(e0.x + d0.x);
      av[1] = (__bf16)fast_tanh(e0.y + d0.y);
      av[2] = (__bf16)fast_tanh(e0.z + d0.z);
      av[3] = (__bf16)fast_tanh(e0.w + d0.w);
      av[4] = (__bf16)fast_tanh(e1.x + d1.x);
      av[5] = (__bf16)fast_tanh(e1.y + d1.y);
      av[6] = (__bf16)fast_tanh(e1.z + d1.z);
      av[7] = (__bf16)fast_tanh(e1.w + d1.w);
    } else {
      const float4* pa = (const float4*)(A0 + (size_t)(m0 + ar) * K_DIM + k0 + as);
      float4 a0 = pa[0], a1 = pa[1];
      av[0] = (__bf16)a0.x; av[1] = (__bf16)a0.y;
      av[2] = (__bf16)a0.z; av[3] = (__bf16)a0.w;
      av[4] = (__bf16)a1.x; av[5] = (__bf16)a1.y;
      av[6] = (__bf16)a1.z; av[7] = (__bf16)a1.w;
    }

    __syncthreads();   // everyone done reading As / Bs[buf^1] from iter it-1
    *(v8bf*)(&As[ar * AST + as]) = av;
    if (it + 1 < NIT) {
      fill_B(k0 + 32, Bs[buf ^ 1]);          // async: overlaps this iter's math
      wait_async<8>();                       // Bs[buf] (older 8 issues) complete
    } else {
      wait_async<0>();
    }
    __syncthreads();   // As + this wave's async portion visible to all

    // ---- WMMA: 2 M-subtiles x 8 N-subtiles per wave ----
    const int r16 = lane & 15;
    const int ka  = (lane >> 4) * 16;   // A frag: 16 contiguous K
    const int kb  = (lane >> 4) * 16;   // B frag: 16 contiguous K

    v16bf afrag[2];
#pragma unroll
    for (int ms = 0; ms < 2; ++ms) {
      const __bf16* p = &As[(wm * 32 + ms * 16 + r16) * AST + ka];
      afrag[ms] = cat16(*(const v8bf*)p, *(const v8bf*)(p + 8));
    }
    const __bf16* Bbase = &Bs[buf][(wn * 128 + r16) * BST + kb];
    v16bf bcur = cat16(*(const v8bf*)Bbase, *(const v8bf*)(Bbase + 8));
#pragma unroll
    for (int ns = 0; ns < 8; ++ns) {
      v16bf bnext = bcur;
      if (ns < 7) {                     // rotate: next frag load hides dscnt wait
        const __bf16* p = Bbase + (ns + 1) * 16 * BST;
        bnext = cat16(*(const v8bf*)p, *(const v8bf*)(p + 8));
      }
      acc[0][ns] = __builtin_amdgcn_wmma_f32_16x16x32_bf16(
          false, afrag[0], false, bcur, (short)0, acc[0][ns], false, false);
      acc[1][ns] = __builtin_amdgcn_wmma_f32_16x16x32_bf16(
          false, afrag[1], false, bcur, (short)0, acc[1][ns], false, false);
      bcur = bnext;
    }
  }

  // ---- epilogue: C layout (ISA 7.12.2): elem v -> M = v + (lane>=16)*8 ----
  const int r16 = lane & 15;
  const int hi8 = (lane >> 4) * 8;
#pragma unroll
  for (int ms = 0; ms < 2; ++ms) {
    const int gm = m0 + wm * 32 + ms * 16 + hi8;
#pragma unroll
    for (int ns = 0; ns < 8; ++ns) {
      const int gn = n0 + wn * 128 + ns * 16 + r16;
      const float bv = bias[gn];
#pragma unroll
      for (int v = 0; v < 8; ++v) {
        Cout[(size_t)(gm + v) * NCOLS + gn] = acc[ms][ns][v] + bv;
      }
    }
  }
}

extern "C" void kernel_launch(void* const* d_in, const int* in_sizes, int n_in,
                              void* d_out, int out_size, void* d_ws, size_t ws_size,
                              hipStream_t stream) {
  const float* enc_out = (const float*)d_in[0];  // (B,T,1,D) = 2048 x 512
  const float* dec_out = (const float*)d_in[1];  // (B,1,U,D) =  512 x 512
  const float* W_enc   = (const float*)d_in[2];  // 512 x 512
  const float* b_enc   = (const float*)d_in[3];
  const float* W_dec   = (const float*)d_in[4];  // 512 x 512
  const float* b_dec   = (const float*)d_in[5];
  const float* W_out   = (const float*)d_in[6];  // 512 x 1024
  const float* b_out   = (const float*)d_in[7];
  float* out = (float*)d_out;                    // (B,T,U,V) = 262144 x 1024

  char* ws = (char*)d_ws;
  float*  enc_proj = (float*)(ws);                           // 4 MiB
  float*  dec_proj = (float*)(ws + (4u << 20));              // 1 MiB
  __bf16* Wt_out   = (__bf16*)(ws + (5u << 20));             // 1 MiB (1024 x 512)
  __bf16* Wt_enc   = (__bf16*)(ws + (6u << 20));             // 512 KiB
  __bf16* Wt_dec   = (__bf16*)(ws + (6u << 20) + (512u << 10));

  // One-shot weight convert+transpose to bf16 (N x K).
  wcvt<<<dim3(2048), 256, 0, stream>>>(W_out, Wt_out, V_DIM);
  wcvt<<<dim3(1024), 256, 0, stream>>>(W_enc, Wt_enc, J_DIM);
  wcvt<<<dim3(1024), 256, 0, stream>>>(W_dec, Wt_dec, J_DIM);

  // Projections (small WMMA GEMMs into workspace).
  joint_gemm<1, 512><<<dim3(32, 1), 256, 0, stream>>>(
      enc_out, nullptr, Wt_enc, b_enc, enc_proj);
  joint_gemm<1, 512><<<dim3(8, 1), 256, 0, stream>>>(
      dec_out, nullptr, Wt_dec, b_dec, dec_proj);

  // Fused tanh-joint x W_out GEMM: M = 262144, N = 1024, K = 512.
  joint_gemm<0, 1024><<<dim3(4096, 2), 256, 0, stream>>>(
      enc_proj, dec_proj, Wt_out, b_out, out);
}
